// ParallelAttentionLSTM_88888643158631
// MI455X (gfx1250) — compile-verified, matrix operand order checked by reading, not run on hardware
//
#include <hip/hip_runtime.h>
#include <hip/hip_bf16.h>

#define USE_ASYNC 1   // gfx1250 async global->LDS tile staging (ASYNCcnt path)

// Problem dims
#define BDIM 32
#define RDIM 4
#define SDIM 32
#define IDIM 512
#define HDIM 1024
#define FDIM 2048
#define NHEAD 8
#define HD 128          // head dim
#define KSEQ 196        // P*P

typedef __bf16 bf16_t;
typedef __attribute__((ext_vector_type(16))) __bf16 v16bf;
typedef __attribute__((ext_vector_type(8)))  __bf16 v8bf;
typedef __attribute__((ext_vector_type(8)))  float  v8f;

// Load a 16-element bf16 WMMA fragment from an LDS row (two 16B chunks).
// Per CDNA5 ISA 16-bit A layout: lane&15 = row-in-tile, lanes 0-15 hold
// K = {0..7, 16..23}, lanes 16-31 hold K = {8..15, 24..31}. B (weights stored
// (N,K) row-major == column-major B) uses the mirrored layout, so the same
// loader serves both operands.
static __device__ __forceinline__ v16bf ld_frag(const bf16_t* rowbase, int kb) {
  v8bf lo = *(const v8bf*)(rowbase + kb);
  v8bf hi = *(const v8bf*)(rowbase + kb + 16);
  return __builtin_shufflevector(lo, hi, 0,1,2,3,4,5,6,7,8,9,10,11,12,13,14,15);
}

// ---------------------------------------------------------------------------
// Generic batched GEMM: for r in [0,gridDim.z):
//   C[r] (MxN, f32) = alpha * (A[r] (MxKd, bf16) * W[r] (NxKd, bf16)^T + bias[r])
//                     (+ C[r] if accumulate)
//   optional bf16 mirror Cbf.
// 8 waves, double-buffered LDS, async global->LDS tile staging,
// v_wmma_f32_16x16x32_bf16 inner product.
// ---------------------------------------------------------------------------
template<int BM, int BN, int WAVES_M, int WAVES_N>
__global__ __launch_bounds__(256)
void gemm_bf16_wmma(const bf16_t* __restrict__ A, long strideA,
                    const bf16_t* __restrict__ W, long strideW,
                    const float* __restrict__ bias, long strideBias,
                    float* __restrict__ C, bf16_t* __restrict__ Cbf, long strideC,
                    int M, int N, int Kd, float alpha, int accumulate) {
  static_assert(WAVES_M * WAVES_N == 8, "8 waves per block");
  constexpr int WMT = BM / (WAVES_M * 16);   // 16x16 M-subtiles per wave
  constexpr int WNT = BN / (WAVES_N * 16);   // 16x16 N-subtiles per wave
  constexpr int LDSROW = 40;                 // bf16/row (80B pitch: 16B-aligned, conflict-free)

  __shared__ bf16_t shA[2][BM * LDSROW];
  __shared__ bf16_t shW[2][BN * LDSROW];

  const int tid  = threadIdx.x;
  const int lane = tid & 31;
  const int wave = tid >> 5;
  const int wm   = wave % WAVES_M;
  const int wn   = wave / WAVES_M;
  const int fr   = lane & 15;          // row within 16x16 tile
  const int kb   = (lane >> 4) * 8;    // K sub-block base (0 or 8)

  const int r  = blockIdx.z;
  const int n0 = blockIdx.x * BN;
  const int m0 = blockIdx.y * BM;
  const bf16_t* Ar = A + (long)r * strideA;
  const bf16_t* Wr = W + (long)r * strideW;
  const float*  Br = bias ? (bias + (long)r * strideBias) : nullptr;
  const long cbase = (long)r * strideC;

  constexpr int CA = (BM * 4 + 255) / 256;   // 16B chunks per thread (A tile)
  constexpr int CW = (BN * 4 + 255) / 256;   // 16B chunks per thread (W tile)

  // Stage one 32-deep K-slab of A and W into LDS buffer `buf`.
  auto copy_tile = [&](int buf, int kk) {
#pragma unroll
    for (int c = 0; c < CA; ++c) {
      const int ch = tid + c * 256;
      if (ch < BM * 4) {
        const int row = ch >> 2, kc = (ch & 3) * 8;
        bf16_t* dst = &shA[buf][row * LDSROW + kc];
        const long goff = ((long)(m0 + row) * Kd + kk + kc) * 2;  // bytes
#if USE_ASYNC
        asm volatile("global_load_async_to_lds_b128 %0, %1, %2 offset:0"
                     :: "v"((unsigned int)(unsigned long)dst),
                        "v"((unsigned int)goff),
                        "s"(Ar)
                     : "memory");
#else
        *(uint4*)dst = *(const uint4*)((const char*)Ar + goff);
#endif
      }
    }
#pragma unroll
    for (int c = 0; c < CW; ++c) {
      const int ch = tid + c * 256;
      if (ch < BN * 4) {
        const int row = ch >> 2, kc = (ch & 3) * 8;
        bf16_t* dst = &shW[buf][row * LDSROW + kc];
        const long goff = ((long)(n0 + row) * Kd + kk + kc) * 2;  // bytes
#if USE_ASYNC
        asm volatile("global_load_async_to_lds_b128 %0, %1, %2 offset:0"
                     :: "v"((unsigned int)(unsigned long)dst),
                        "v"((unsigned int)goff),
                        "s"(Wr)
                     : "memory");
#else
        *(uint4*)dst = *(const uint4*)((const char*)Wr + goff);
#endif
      }
    }
  };

  v8f acc[WMT][WNT];
  const v8f vzero = {0.f, 0.f, 0.f, 0.f, 0.f, 0.f, 0.f, 0.f};
#pragma unroll
  for (int i = 0; i < WMT; ++i)
#pragma unroll
    for (int j = 0; j < WNT; ++j) acc[i][j] = vzero;

  // Prologue: stage slab 0.
  copy_tile(0, 0);
#if USE_ASYNC
  asm volatile("s_wait_asynccnt 0" ::: "memory");
#endif
  __syncthreads();

  const int nsteps = Kd >> 5;
  for (int s = 0; s < nsteps; ++s) {
    const int buf = s & 1;

#if USE_ASYNC
    // Issue async staging of the next slab; it lands in LDS while we compute.
    if (s + 1 < nsteps) copy_tile(buf ^ 1, (s + 1) * 32);
#endif
    if (s + 2 < nsteps) {
      __builtin_prefetch(Ar + (long)(m0 + (tid % BM)) * Kd + (s + 2) * 32, 0, 1);
      __builtin_prefetch(Wr + (long)(n0 + (tid % BN)) * Kd + (s + 2) * 32, 0, 1);
    }

    v16bf af[WMT], wf[WNT];
#pragma unroll
    for (int i = 0; i < WMT; ++i) {
      const int row = (wm * WMT + i) * 16 + fr;
      af[i] = ld_frag(&shA[buf][row * LDSROW], kb);
    }
#pragma unroll
    for (int j = 0; j < WNT; ++j) {
      const int row = (wn * WNT + j) * 16 + fr;
      wf[j] = ld_frag(&shW[buf][row * LDSROW], kb);
    }
#pragma unroll
    for (int i = 0; i < WMT; ++i)
#pragma unroll
      for (int j = 0; j < WNT; ++j)
        acc[i][j] = __builtin_amdgcn_wmma_f32_16x16x32_bf16(
            false, af[i], false, wf[j], (short)0, acc[i][j], false, false);

#if USE_ASYNC
    asm volatile("s_wait_asynccnt 0" ::: "memory");
#else
    // Sync fallback: stage next slab after compute so fragment ds_loads are
    // not queued behind the tile ds_stores.
    if (s + 1 < nsteps) copy_tile(buf ^ 1, (s + 1) * 32);
#endif
    __syncthreads();
  }

  // Epilogue. C/D layout: lane&15 = N, (lane>>4)*8 + vgpr = M within tile.
#pragma unroll
  for (int i = 0; i < WMT; ++i) {
    const int cmb = m0 + (wm * WMT + i) * 16 + (lane >> 4) * 8;
#pragma unroll
    for (int j = 0; j < WNT; ++j) {
      const int cn = n0 + (wn * WNT + j) * 16 + (lane & 15);
      const float bv = Br ? Br[cn] : 0.f;
#pragma unroll
      for (int e = 0; e < 8; ++e) {
        const long idx = cbase + (long)(cmb + e) * N + cn;
        float v = alpha * (acc[i][j][e] + bv);
        if (accumulate) v += C[idx];
        if (C)   C[idx]   = v;
        if (Cbf) Cbf[idx] = (bf16_t)v;
      }
    }
  }
}

// ---------------------------------------------------------------------------
// Single-query MHA: one block per (head, b, r). Scores over K=196 (VALU dot,
// tiny FLOPs), block softmax, context over K, bf16 output for the o-proj GEMM.
// ---------------------------------------------------------------------------
__global__ __launch_bounds__(256)
void attn_kernel(const float* __restrict__ q, const bf16_t* __restrict__ kp,
                 const bf16_t* __restrict__ vp, bf16_t* __restrict__ ctx) {
  const int n = blockIdx.x, b = blockIdx.y, r = blockIdx.z;
  const int tid = threadIdx.x;
  __shared__ float sq[HD];
  __shared__ float sp[256];
  __shared__ float red[256];

  const long qoff = ((long)r * BDIM + b) * HDIM + (long)n * HD;
  if (tid < HD) sq[tid] = q[qoff + tid];
  __syncthreads();

  float s = -3.0e38f;
  if (tid < KSEQ) {
    const bf16_t* kr = kp + (((long)r * BDIM + b) * KSEQ + tid) * HDIM + (long)n * HD;
    float a = 0.f;
#pragma unroll 8
    for (int d = 0; d < HD; ++d) a += sq[d] * (float)kr[d];
    s = a;
  }
  red[tid] = s; __syncthreads();
  for (int o = 128; o > 0; o >>= 1) { if (tid < o) red[tid] = fmaxf(red[tid], red[tid + o]); __syncthreads(); }
  const float mx = red[0]; __syncthreads();
  const float e = (tid < KSEQ) ? __expf(s - mx) : 0.f;
  sp[tid] = e;
  red[tid] = e; __syncthreads();
  for (int o = 128; o > 0; o >>= 1) { if (tid < o) red[tid] += red[tid + o]; __syncthreads(); }
  const float inv = 1.f / red[0];
  __syncthreads();

  if (tid < HD) {
    const bf16_t* vr = vp + (((long)r * BDIM + b) * KSEQ) * HDIM + (long)n * HD + tid;
    float a = 0.f;
    for (int k = 0; k < KSEQ; ++k) a += sp[k] * (float)vr[(long)k * HDIM];
    ctx[qoff + tid] = (bf16_t)(a * inv);
  }
}

// LSTM pointwise cell: gates (i,f,g,o) -> h_new; c carry is replaced by the
// attention context afterwards (per reference), so c_new stays local.
__global__ void lstm_cell_kernel(const float* __restrict__ g, const float* __restrict__ c,
                                 bf16_t* __restrict__ hbf, float* __restrict__ out,
                                 float* __restrict__ hn, int t) {
  const long idx = (long)blockIdx.x * blockDim.x + threadIdx.x;  // < R*B*H
  const int hh = (int)(idx % HDIM);
  const int b  = (int)((idx / HDIM) % BDIM);
  const int r  = (int)(idx / ((long)HDIM * BDIM));
  const float* gr = g + ((long)r * BDIM + b) * 4 * HDIM;
  const float ig = 1.f / (1.f + __expf(-gr[hh]));
  const float fg = 1.f / (1.f + __expf(-gr[HDIM + hh]));
  const float gg = tanhf(gr[2 * HDIM + hh]);
  const float og = 1.f / (1.f + __expf(-gr[3 * HDIM + hh]));
  const float cv = fg * c[idx] + ig * gg;
  const float hv = og * tanhf(cv);
  hbf[idx] = (bf16_t)hv;
  out[(((long)r * BDIM + b) * SDIM + t) * HDIM + hh] = hv;
  if (hn) hn[idx] = hv;
}

// ------------------------------- prep kernels ------------------------------
__global__ void cvt_kernel(const float* __restrict__ s, bf16_t* __restrict__ d, long n) {
  for (long i = (long)blockIdx.x * blockDim.x + threadIdx.x; i < n; i += (long)gridDim.x * blockDim.x)
    d[i] = (bf16_t)s[i];
}

__global__ void pool_kernel(const float* __restrict__ feat, float* __restrict__ pooled,
                            bf16_t* __restrict__ pbf) {
  const int idx = blockIdx.x * blockDim.x + threadIdx.x;  // B*F
  if (idx >= BDIM * FDIM) return;
  const float* p = feat + (long)idx * KSEQ;
  float a = 0.f;
  for (int k = 0; k < KSEQ; ++k) a += p[k];
  a *= (1.f / (float)KSEQ);
  pooled[idx] = a;
  pbf[idx] = (bf16_t)a;
}

__global__ void keyseq_kernel(const float* __restrict__ feat, bf16_t* __restrict__ ks) {
  const long total = (long)BDIM * KSEQ * FDIM;
  for (long i = (long)blockIdx.x * blockDim.x + threadIdx.x; i < total; i += (long)gridDim.x * blockDim.x) {
    const int  f  = (int)(i % FDIM);
    const long bk = i / FDIM;
    const int  k  = (int)(bk % KSEQ);
    const long b  = bk / KSEQ;
    ks[i] = (bf16_t)feat[(b * FDIM + f) * KSEQ + k];
  }
}

__global__ void xprep_kernel(const float* __restrict__ wds, bf16_t* __restrict__ x) {
  const long total = (long)RDIM * SDIM * BDIM * IDIM;
  for (long i = (long)blockIdx.x * blockDim.x + threadIdx.x; i < total; i += (long)gridDim.x * blockDim.x) {
    const int ii = (int)(i % IDIM);
    const int b  = (int)((i / IDIM) % BDIM);
    const int s  = (int)((i / ((long)IDIM * BDIM)) % SDIM);
    const int r  = (int)(i / ((long)IDIM * BDIM * SDIM));
    x[i] = (bf16_t)wds[(((long)b * RDIM + r) * SDIM + s) * IDIM + ii];
  }
}

__global__ void copy_kernel(const float* __restrict__ s, float* __restrict__ d, long n) {
  for (long i = (long)blockIdx.x * blockDim.x + threadIdx.x; i < n; i += (long)gridDim.x * blockDim.x)
    d[i] = s[i];
}

// ---------------------------------------------------------------------------
extern "C" void kernel_launch(void* const* d_in, const int* in_sizes, int n_in,
                              void* d_out, int out_size, void* d_ws, size_t ws_size,
                              hipStream_t stream) {
  (void)in_sizes; (void)n_in; (void)out_size; (void)ws_size;

  const float* wds  = (const float*)d_in[0];
  const float* feat = (const float*)d_in[1];
  const float* Wi   = (const float*)d_in[2];
  const float* bi   = (const float*)d_in[3];
  const float* qw   = (const float*)d_in[4];
  const float* qb   = (const float*)d_in[5];
  const float* kw   = (const float*)d_in[6];
  const float* kbv  = (const float*)d_in[7];
  const float* vw   = (const float*)d_in[8];
  const float* vb   = (const float*)d_in[9];
  const float* ow   = (const float*)d_in[10];
  const float* ob   = (const float*)d_in[11];
  const float* Wih  = (const float*)d_in[12];
  const float* Whh  = (const float*)d_in[13];
  const float* bih  = (const float*)d_in[14];
  const float* bhh  = (const float*)d_in[15];

  char* p = (char*)d_ws;
  auto alloc = [&](size_t bytes) -> void* {
    void* q = (void*)p;
    p += (bytes + 255) & ~(size_t)255;
    return q;
  };

  const long HF   = (long)HDIM * FDIM;
  const long HH   = (long)HDIM * HDIM;
  const long H4I  = (long)4 * HDIM * IDIM;
  const long H4H  = (long)4 * HDIM * HDIM;
  const long RHF  = RDIM * HF;
  const long RHH  = RDIM * HH;
  const long R4HI = RDIM * H4I;
  const long R4HH = RDIM * H4H;
  const long BKF  = (long)BDIM * KSEQ * FDIM;
  const long RSBI = (long)RDIM * SDIM * BDIM * IDIM;
  const long BKH  = (long)BDIM * KSEQ * HDIM;
  const long RBKH = RDIM * BKH;
  const long BH   = (long)BDIM * HDIM;
  const long RBH  = RDIM * BH;

  bf16_t* Wi_b  = (bf16_t*)alloc(RHF * 2);
  bf16_t* qw_b  = (bf16_t*)alloc(RHH * 2);
  bf16_t* kw_b  = (bf16_t*)alloc(RHF * 2);
  bf16_t* vw_b  = (bf16_t*)alloc(RHF * 2);
  bf16_t* ow_b  = (bf16_t*)alloc(RHH * 2);
  bf16_t* Wih_b = (bf16_t*)alloc(R4HI * 2);
  bf16_t* Whh_b = (bf16_t*)alloc(R4HH * 2);
  bf16_t* ksq   = (bf16_t*)alloc(BKF * 2);
  bf16_t* xb    = (bf16_t*)alloc(RSBI * 2);
  bf16_t* kp    = (bf16_t*)alloc(RBKH * 2);
  bf16_t* vp    = (bf16_t*)alloc(RBKH * 2);
  bf16_t* hbf   = (bf16_t*)alloc(RBH * 2);
  bf16_t* ctxb  = (bf16_t*)alloc(RBH * 2);
  bf16_t* poolb = (bf16_t*)alloc((long)BDIM * FDIM * 2);
  float*  poolf = (float*)alloc((long)BDIM * FDIM * 4);
  float*  qf    = (float*)alloc(RBH * 4);
  float*  gbuf  = (float*)alloc((long)RDIM * BDIM * 4 * HDIM * 4);
  float*  cbuf  = (float*)alloc(RBH * 4);

  float* out = (float*)d_out;
  float* hn  = out + (long)RDIM * BDIM * SDIM * HDIM;
  float* cn  = hn + RBH;

  // --- prep: bf16 conversions + layout changes ---
  cvt_kernel<<<2048, 256, 0, stream>>>(Wi,  Wi_b,  RHF);
  cvt_kernel<<<2048, 256, 0, stream>>>(qw,  qw_b,  RHH);
  cvt_kernel<<<2048, 256, 0, stream>>>(kw,  kw_b,  RHF);
  cvt_kernel<<<2048, 256, 0, stream>>>(vw,  vw_b,  RHF);
  cvt_kernel<<<2048, 256, 0, stream>>>(ow,  ow_b,  RHH);
  cvt_kernel<<<2048, 256, 0, stream>>>(Wih, Wih_b, R4HI);
  cvt_kernel<<<4096, 256, 0, stream>>>(Whh, Whh_b, R4HH);
  pool_kernel<<<(BDIM * FDIM + 255) / 256, 256, 0, stream>>>(feat, poolf, poolb);
  keyseq_kernel<<<4096, 256, 0, stream>>>(feat, ksq);
  xprep_kernel<<<2048, 256, 0, stream>>>(wds, xb);

  const float scale = 0.08838834764831845f;  // 1/sqrt(HD)

  // --- time-invariant K/V projections: (6272 x 1024) = (6272 x 2048) * W^T ---
  gemm_bf16_wmma<128, 128, 4, 2><<<dim3(HDIM / 128, (BDIM * KSEQ) / 128, RDIM), 256, 0, stream>>>(
      ksq, 0, kw_b, HF, kbv, HDIM, nullptr, kp, BKH, BDIM * KSEQ, HDIM, FDIM, 1.f, 0);
  gemm_bf16_wmma<128, 128, 4, 2><<<dim3(HDIM / 128, (BDIM * KSEQ) / 128, RDIM), 256, 0, stream>>>(
      ksq, 0, vw_b, HF, vb, HDIM, nullptr, vp, BKH, BDIM * KSEQ, HDIM, FDIM, 1.f, 0);

  // --- h0 = pooled @ Wi^T + bi ---
  gemm_bf16_wmma<32, 256, 1, 8><<<dim3(HDIM / 256, 1, RDIM), 256, 0, stream>>>(
      poolb, 0, Wi_b, HF, bi, HDIM, nullptr, hbf, BH, BDIM, HDIM, FDIM, 1.f, 0);

  auto q_gemm = [&]() {
    gemm_bf16_wmma<32, 256, 1, 8><<<dim3(HDIM / 256, 1, RDIM), 256, 0, stream>>>(
        hbf, BH, qw_b, HH, qb, HDIM, qf, nullptr, BH, BDIM, HDIM, HDIM, scale, 0);
  };
  auto attn = [&]() {
    attn_kernel<<<dim3(NHEAD, BDIM, RDIM), 256, 0, stream>>>(qf, kp, vp, ctxb);
  };
  auto o_gemm = [&]() {
    gemm_bf16_wmma<32, 256, 1, 8><<<dim3(HDIM / 256, 1, RDIM), 256, 0, stream>>>(
        ctxb, BH, ow_b, HH, ob, HDIM, cbuf, nullptr, BH, BDIM, HDIM, HDIM, 1.f, 0);
  };

  // c0 = attend(h0)
  q_gemm(); attn(); o_gemm();

  // --- recurrent loop ---
  for (int t = 0; t < SDIM; ++t) {
    // gates = x_t @ Wih^T + b_ih  (then += h @ Whh^T + b_hh)
    gemm_bf16_wmma<32, 256, 1, 8><<<dim3(4 * HDIM / 256, 1, RDIM), 256, 0, stream>>>(
        xb + (long)t * BDIM * IDIM, (long)SDIM * BDIM * IDIM, Wih_b, H4I, bih, 4 * HDIM,
        gbuf, nullptr, (long)BDIM * 4 * HDIM, BDIM, 4 * HDIM, IDIM, 1.f, 0);
    gemm_bf16_wmma<32, 256, 1, 8><<<dim3(4 * HDIM / 256, 1, RDIM), 256, 0, stream>>>(
        hbf, BH, Whh_b, H4H, bhh, 4 * HDIM,
        gbuf, nullptr, (long)BDIM * 4 * HDIM, BDIM, 4 * HDIM, HDIM, 1.f, 1);
    lstm_cell_kernel<<<(int)(RBH / 256), 256, 0, stream>>>(
        gbuf, cbuf, hbf, out, (t == SDIM - 1) ? hn : nullptr, t);
    // c <- attend(h_new)
    q_gemm(); attn(); o_gemm();
  }

  copy_kernel<<<512, 256, 0, stream>>>(cbuf, cn, RBH);
}
